// FSAGRUScorer_68599217652222
// MI455X (gfx1250) — compile-verified, matrix operand order checked by reading, not run on hardware
//
#include <hip/hip_runtime.h>
#include <hip/hip_bf16.h>

#define BB 128
#define TT 512
#define VV 512
#define HH 256
#define CC 256
#define H3 768
#define PAD_TOK 0
#define BOS_TOK 1
#define EOS_TOK 2

typedef __bf16 bf16;
typedef bf16 v16bf __attribute__((ext_vector_type(16)));
typedef float v8f  __attribute__((ext_vector_type(8)));
typedef unsigned int u32x4 __attribute__((ext_vector_type(4)));

// TDM descriptor SGPR-group types (D# group0=4, group1=8, group2/3=4 dwords)
typedef unsigned int tdm_u32x4 __attribute__((ext_vector_type(4)));
typedef int          tdm_i32x4 __attribute__((ext_vector_type(4)));
typedef int          tdm_i32x8 __attribute__((ext_vector_type(8)));

#if defined(__HIP_DEVICE_COMPILE__) && __has_builtin(__builtin_amdgcn_tensor_load_to_lds) && \
    __has_builtin(__builtin_amdgcn_s_wait_tensorcnt)
#define HAVE_TDM 1
#else
#define HAVE_TDM 0
#endif

union FragAB { v16bf v; u32x4 q[2]; };

// Load a 16x16x32 WMMA A/B fragment (16 bf16 per lane = two 16B chunks).
// ISA layout: lane L -> row/col (L&15), K chunks at kb+8*(L>>4) and kb+16+8*(L>>4).
__device__ __forceinline__ v16bf load_frag(const bf16* row, int kb, int half) {
  FragAB f;
  f.q[0] = *reinterpret_cast<const u32x4*>(row + kb + 8 * half);
  f.q[1] = *reinterpret_cast<const u32x4*>(row + kb + 16 + 8 * half);
  return f.v;
}

__device__ __forceinline__ v8f wmma_bf16(v16bf a, v16bf b, v8f c) {
  return __builtin_amdgcn_wmma_f32_16x16x32_bf16(
      /*neg_a=*/false, a, /*neg_b=*/false, b,
      /*c_mod=*/(short)0, c, /*reuse_a=*/false, /*reuse_b=*/false);
}

__device__ __forceinline__ float sigmoidf_(float x) { return 1.f / (1.f + __expf(-x)); }

// TDM: async DMA of a 16x256 f32 context tile (row stride T*C) into LDS.
// D# per CDNA5 ISA ch.8: group0 = {flags, lds_addr, global_addr, type=2},
// group1 = {mask/data_size, tensor_dim0/1, tile_dim0/1/2, dim0_stride}.
__device__ __forceinline__ void tdm_load_ctx_tile(const float* gsrc, float* lds_dst) {
#if HAVE_TDM
  unsigned long long ga = (unsigned long long)(uintptr_t)gsrc;
  unsigned int lds_off  = (unsigned int)(uintptr_t)lds_dst;  // generic addr low 32b = LDS byte addr
  tdm_u32x4 g0;
  g0[0] = 1u;                                             // count=1, user mode, no gather
  g0[1] = lds_off;                                        // lds_addr
  g0[2] = (unsigned int)(ga & 0xFFFFFFFFu);               // global_addr[31:0]
  g0[3] = (unsigned int)((ga >> 32) & 0x01FFFFFFu)        // global_addr[56:32]
          | (2u << 30);                                   // type = 2 ("image")
  tdm_i32x8 g1;
  g1[0] = (2 << 16);                                      // data_size=2 -> 4-byte elems; mask=0
  g1[1] = (int)(256u << 16);                              // tensor_dim0 = 256 (bits 79:48 lo)
  g1[2] = (int)(16u << 16);                               // tensor_dim1 = 16  (bits 111:80 lo)
  g1[3] = (int)(256u << 16);                              // tile_dim0 = 256   (bits 127:112)
  g1[4] = 16;                                             // tile_dim1 = 16; tile_dim2 = 0
  g1[5] = (int)(TT * CC);                                 // tensor_dim0_stride = 131072 elems
  g1[6] = 0;
  g1[7] = 0;
  tdm_i32x4 gz = {0, 0, 0, 0};
#if __clang_major__ >= 23
  tdm_i32x8 gz8 = {0, 0, 0, 0, 0, 0, 0, 0};
  __builtin_amdgcn_tensor_load_to_lds(g0, g1, gz, gz, gz8, 0);
#else
  __builtin_amdgcn_tensor_load_to_lds(g0, g1, gz, gz, 0);
#endif
#endif
}

// ---------------------------------------------------------------------------
// Kernel 0: convert fp32 weights/embeddings to bf16 into workspace.
// ws layout (elements): emb[131072] | Wih[196608] | Whh[196608] | Wc[131072] | Wo[131072]
// ---------------------------------------------------------------------------
__global__ void convert_weights(const float* __restrict__ emb,
                                const float* __restrict__ Wih,
                                const float* __restrict__ Whh,
                                const float* __restrict__ Wc,
                                const float* __restrict__ Wo,
                                bf16* __restrict__ ws) {
  const int n_emb = VV * HH;          // 131072
  const int n_w   = H3 * HH;          // 196608
  const int n_wc  = HH * (CC + HH);   // 131072
  const int n_wo  = VV * HH;          // 131072
  const int total = n_emb + 2 * n_w + n_wc + n_wo;
  for (int i = blockIdx.x * blockDim.x + threadIdx.x; i < total;
       i += gridDim.x * blockDim.x) {
    float v;
    if      (i < n_emb)                   v = emb[i];
    else if (i < n_emb + n_w)             v = Wih[i - n_emb];
    else if (i < n_emb + 2 * n_w)         v = Whh[i - n_emb - n_w];
    else if (i < n_emb + 2 * n_w + n_wc)  v = Wc[i - n_emb - 2 * n_w];
    else                                  v = Wo[i - n_emb - 2 * n_w - n_wc];
    ws[i] = (bf16)v;
  }
}

// ---------------------------------------------------------------------------
// Kernel 1: persistent GRU scan. One WG per 16 batch rows, 8 wave32s.
// Per step: TDM prefetch of ctx tile overlaps [x|h]@[Wih;Whh]^T (r,z, K=512)
// + n-gate reg GEMMs -> GRU elementwise -> [q|h_new]@Wc^T -> hc@Wo^T ->
// masked log-softmax (cross-wave reduction).
// ---------------------------------------------------------------------------
__global__ __launch_bounds__(256)
void gru_scan_kernel(const int* __restrict__ seq, const float* __restrict__ ctx,
                     const bf16* __restrict__ emb_bf, const bf16* __restrict__ Wih_bf,
                     const bf16* __restrict__ Whh_bf, const bf16* __restrict__ Wc_bf,
                     const bf16* __restrict__ Wo_bf,
                     const float* __restrict__ b_ih, const float* __restrict__ b_hh,
                     const float* __restrict__ b_c,  const float* __restrict__ b_o,
                     float* __restrict__ out) {
  __shared__ __attribute__((aligned(16))) bf16  smemA[16][512]; // [x|h] then [q|h_new]
  __shared__ __attribute__((aligned(16))) float G[16][512];     // r,z preact; reused as hc bf16
  __shared__ __attribute__((aligned(16))) float qstage[16][256]; // TDM landing pad (f32 ctx tile)
  __shared__ float red_max[8][16];
  __shared__ float red_sum[8][16];
  __shared__ float rowmax[16];
  __shared__ float tgtlog[16];
  __shared__ int   prevtok_s[16];
  __shared__ int   tgt_s[16];

  bf16* hc_bf = reinterpret_cast<bf16*>(&G[0][0]); // 16x256 bf16, aliases G

  const int tid  = threadIdx.x;
  const int wave = tid >> 5;
  const int lane = tid & 31;
  const int half = lane >> 4;
  const int lc   = lane & 15;
  const int b0   = blockIdx.x * 16;

  float acc = 0.f; // sequence log-score, owned by threads 0..15

  // h = 0 (upper half of smemA persistently holds h in bf16)
  for (int i = tid; i < 16 * 256; i += 256)
    smemA[i >> 8][256 + (i & 255)] = (bf16)0.f;
  __syncthreads();

  for (int t = 0; t < TT; ++t) {
    // ---- Phase A: TDM ctx prefetch + tokens + embed lookup --------------
    if (wave == 0) {
      // async DMA: context[b0..b0+15, t, :] -> qstage, overlapped with GEMM1/2
      tdm_load_ctx_tile(ctx + ((size_t)b0 * TT + t) * CC, &qstage[0][0]);
    }
    if (tid < 16) {
      const size_t row = (size_t)(b0 + tid) * TT;
      prevtok_s[tid] = (t == 0) ? BOS_TOK : seq[row + t - 1];
      tgt_s[tid]     = seq[row + t];
      tgtlog[tid]    = -1e30f;
      __builtin_prefetch(ctx + (row + ((t + 1 < TT) ? t + 1 : t)) * CC, 0, 1);
    }
    __syncthreads();
    {
      const int m  = tid >> 4;
      const int c0 = (tid & 15) * 16;
      const bf16* src = emb_bf + (size_t)prevtok_s[m] * HH + c0;
      u32x4* dst = reinterpret_cast<u32x4*>(&smemA[m][c0]);
      dst[0] = reinterpret_cast<const u32x4*>(src)[0];
      dst[1] = reinterpret_cast<const u32x4*>(src)[1];
    }
    __syncthreads();

    // ---- Phase B: GEMM1 r,z gates: G[16][512] = [x|h] @ [Wih;Whh]^T -----
    for (int tb = 0; tb < 4; ++tb) {
      const int n0 = wave * 64 + tb * 16;
      v8f c = {0.f, 0.f, 0.f, 0.f, 0.f, 0.f, 0.f, 0.f};
      for (int kb = 0; kb < 512; kb += 32) {
        v16bf a = load_frag(&smemA[lc][0], kb, half);
        const bf16* wrow = (kb < 256)
            ? (Wih_bf + (size_t)(n0 + lc) * HH + kb)
            : (Whh_bf + (size_t)(n0 + lc) * HH + (kb - 256));
        c = wmma_bf16(a, load_frag(wrow, 0, half), c);
      }
      const float bias = b_ih[n0 + lc] + b_hh[n0 + lc];
      for (int r = 0; r < 8; ++r) G[r + 8 * half][n0 + lc] = c[r] + bias;
    }

    // ---- GEMM2: n-gate xn/hn kept in registers (2 tiles per wave) -------
    v8f xn[2], hn[2];
    for (int tb = 0; tb < 2; ++tb) {
      const int n0 = wave * 32 + tb * 16;
      v8f cx = {0.f, 0.f, 0.f, 0.f, 0.f, 0.f, 0.f, 0.f};
      v8f ch = {0.f, 0.f, 0.f, 0.f, 0.f, 0.f, 0.f, 0.f};
      for (int kb = 0; kb < 256; kb += 32) {
        v16bf ax = load_frag(&smemA[lc][0],   kb, half);
        v16bf ah = load_frag(&smemA[lc][256], kb, half);
        const bf16* wx = Wih_bf + (size_t)(512 + n0 + lc) * HH + kb;
        const bf16* wh = Whh_bf + (size_t)(512 + n0 + lc) * HH + kb;
        cx = wmma_bf16(ax, load_frag(wx, 0, half), cx);
        ch = wmma_bf16(ah, load_frag(wh, 0, half), ch);
      }
      xn[tb] = cx; hn[tb] = ch;
    }
#if HAVE_TDM
    if (wave == 0) __builtin_amdgcn_s_wait_tensorcnt(0); // TENSORcnt is per-wave
#endif
    __syncthreads(); // publishes TDM-written qstage to all waves

    // ---- Phase C: GRU elementwise (own tile coords) + q -> bf16 ---------
    for (int tb = 0; tb < 2; ++tb) {
      const int c   = wave * 32 + tb * 16 + lc;
      const float bxn = b_ih[512 + c], bhn = b_hh[512 + c];
      for (int r = 0; r < 8; ++r) {
        const int m = r + 8 * half;
        const float rv = sigmoidf_(G[m][c]);
        const float zv = sigmoidf_(G[m][256 + c]);
        const float nn = tanhf(xn[tb][r] + bxn + rv * (hn[tb][r] + bhn));
        const float ho = (float)smemA[m][256 + c];
        float hnew = (1.f - zv) * nn + zv * ho;
        if (prevtok_s[m] == PAD_TOK) hnew = ho;  // hidden_state_masking
        smemA[m][256 + c] = (bf16)hnew;
      }
    }
    {
      const int m  = tid >> 4;
      const int c0 = (tid & 15) * 16;
#if HAVE_TDM
      const float* src = &qstage[m][c0];
#else
      const float* src = ctx + ((size_t)(b0 + m) * TT + t) * CC + c0;
#endif
      for (int j = 0; j < 16; ++j) smemA[m][c0 + j] = (bf16)src[j];
    }
    __syncthreads();

    // ---- Phase D: GEMM3 hc = tanh([q|h_new] @ Wc^T + b_c) ---------------
    for (int tb = 0; tb < 2; ++tb) {
      const int n0 = wave * 32 + tb * 16;
      v8f c = {0.f, 0.f, 0.f, 0.f, 0.f, 0.f, 0.f, 0.f};
      for (int kb = 0; kb < 512; kb += 32) {
        v16bf a = load_frag(&smemA[lc][0], kb, half);
        const bf16* wr = Wc_bf + (size_t)(n0 + lc) * (CC + HH) + kb;
        c = wmma_bf16(a, load_frag(wr, 0, half), c);
      }
      const float bias = b_c[n0 + lc];
      for (int r = 0; r < 8; ++r)
        hc_bf[(size_t)(r + 8 * half) * 256 + (n0 + lc)] = (bf16)tanhf(c[r] + bias);
    }
    __syncthreads();

    // ---- Phase E: GEMM4 scores + masks (4 tiles per wave, in regs) ------
    v8f sc[4];
    for (int tb = 0; tb < 4; ++tb) {
      const int n0 = wave * 64 + tb * 16;
      v8f c = {0.f, 0.f, 0.f, 0.f, 0.f, 0.f, 0.f, 0.f};
      for (int kb = 0; kb < 256; kb += 32) {
        v16bf a = load_frag(hc_bf + (size_t)lc * 256, kb, half);
        const bf16* wr = Wo_bf + (size_t)(n0 + lc) * HH + kb;
        c = wmma_bf16(a, load_frag(wr, 0, half), c);
      }
      const int col = n0 + lc;                   // vocab index
      const float bias = b_o[col];
      for (int r = 0; r < 8; ++r) {
        const int m = r + 8 * half;
        const bool is_ep = (prevtok_s[m] == EOS_TOK) || (prevtok_s[m] == PAD_TOK);
        float logit;
        if (is_ep) logit = (col == PAD_TOK) ? 0.f : -1e30f;
        else       logit = (col == PAD_TOK || col == BOS_TOK) ? -1e30f : (c[r] + bias);
        c[r] = logit;
        if (col == tgt_s[m]) tgtlog[m] = logit;  // unique writer per row
      }
      sc[tb] = c;
    }

    // row-max over this wave's 64 columns (xor-shuffle within 16-lane halves)
    for (int r = 0; r < 8; ++r) {
      float mv = fmaxf(fmaxf(sc[0][r], sc[1][r]), fmaxf(sc[2][r], sc[3][r]));
      for (int msk = 1; msk < 16; msk <<= 1) mv = fmaxf(mv, __shfl_xor(mv, msk, 32));
      if (lc == 0) red_max[wave][r + 8 * half] = mv;
    }
    __syncthreads();
    if (tid < 16) {
      float mv = red_max[0][tid];
      for (int w = 1; w < 8; ++w) mv = fmaxf(mv, red_max[w][tid]);
      rowmax[tid] = mv;
    }
    __syncthreads();
    for (int r = 0; r < 8; ++r) {
      const float mrow = rowmax[r + 8 * half];
      float s = __expf(sc[0][r] - mrow) + __expf(sc[1][r] - mrow) +
                __expf(sc[2][r] - mrow) + __expf(sc[3][r] - mrow);
      for (int msk = 1; msk < 16; msk <<= 1) s += __shfl_xor(s, msk, 32);
      if (lc == 0) red_sum[wave][r + 8 * half] = s;
    }
    __syncthreads();
    if (tid < 16) {
      float s = 0.f;
      for (int w = 0; w < 8; ++w) s += red_sum[w][tid];
      const float lp = tgtlog[tid] - rowmax[tid] - __logf(s);
      if (tgt_s[tid] != PAD_TOK) acc += lp;
    }
    __syncthreads();
  }

  if (tid < 16) out[b0 + tid] = acc;
}

// ---------------------------------------------------------------------------
extern "C" void kernel_launch(void* const* d_in, const int* in_sizes, int n_in,
                              void* d_out, int out_size, void* d_ws, size_t ws_size,
                              hipStream_t stream) {
  const int*   seq = (const int*)d_in[0];
  const float* ctx = (const float*)d_in[1];
  const float* emb = (const float*)d_in[2];
  const float* Wih = (const float*)d_in[3];
  const float* Whh = (const float*)d_in[4];
  const float* bih = (const float*)d_in[5];
  const float* bhh = (const float*)d_in[6];
  const float* Wc  = (const float*)d_in[7];
  const float* bc  = (const float*)d_in[8];
  const float* Wo  = (const float*)d_in[9];
  const float* bo  = (const float*)d_in[10];
  float* out = (float*)d_out;

  bf16* wsb = (bf16*)d_ws;
  convert_weights<<<768, 256, 0, stream>>>(emb, Wih, Whh, Wc, Wo, wsb);

  const bf16* emb_bf = wsb;
  const bf16* Wih_bf = wsb + 131072;
  const bf16* Whh_bf = wsb + 327680;
  const bf16* Wc_bf  = wsb + 524288;
  const bf16* Wo_bf  = wsb + 655360;

  gru_scan_kernel<<<BB / 16, 256, 0, stream>>>(seq, ctx, emb_bf, Wih_bf, Whh_bf,
                                               Wc_bf, Wo_bf, bih, bhh, bc, bo, out);
}
